// MultiHeadAttention_592705487093
// MI455X (gfx1250) — matrix operand order repaired and checked by reading, not compiled, not run on hardware
//
#include <hip/hip_runtime.h>

#define USE_ASYNC_LDS 1   // gfx1250 global_load_async_to_lds_b128 path

// ---------------------------------------------------------------------------
// Types for CDNA5 WMMA (gfx1250, wave32)
// ---------------------------------------------------------------------------
typedef unsigned short u16_t;
typedef __attribute__((ext_vector_type(16))) __bf16 v16bf;
typedef __attribute__((ext_vector_type(8)))  float  v8f;

union Frag {
    uint4  q[2];
    v16bf  v;
    u16_t  us[16];
};

// Problem constants (reference: B=8, S=1024, C=1024, H=16, D=64)
#define BN 8
#define SN 1024
#define CN 1024
#define HN 16
#define DN 64
#define KDIM 1024          // inner dim of both GEMMs
#define NKV  2048          // K+V slices only (Q slice is dead: q = k = qkv[1])
#define MROWS 8192         // B*S

__device__ __forceinline__ u16_t f2bf(float f) {
    unsigned int u = __float_as_uint(f);
    unsigned int r = u + 0x7fffu + ((u >> 16) & 1u);   // round-to-nearest-even
    return (u16_t)(r >> 16);
}

// A-fragment (16x32 bf16, row-major source, ld in elements).
// ISA layout: lane(m = l&15, hi = l>>4), element i -> k = (i>>3)*16 + hi*8 + (i&7)
__device__ __forceinline__ Frag load_a(const u16_t* p, int ld, int lane) {
    int m = lane & 15, hi = lane >> 4;
    const u16_t* r = p + (size_t)m * ld + hi * 8;
    Frag f;
    f.q[0] = *(const uint4*)(r);
    f.q[1] = *(const uint4*)(r + 16);
    return f;
}

// B-fragment (32x16 bf16) loaded from a row-major [N][K] (i.e. B^T) source.
// ISA layout: lane(n = l&15, hi = l>>4), element i -> k = (i>>3)*16 + hi*8 + (i&7)
__device__ __forceinline__ Frag load_bt(const u16_t* p, int ld, int lane) {
    int n = lane & 15, hi = lane >> 4;
    const u16_t* r = p + (size_t)n * ld + hi * 8;
    Frag f;
    f.q[0] = *(const uint4*)(r);
    f.q[1] = *(const uint4*)(r + 16);
    return f;
}

__device__ __forceinline__ v8f wmma_bf16(const Frag& a, const Frag& b, v8f c) {
    return __builtin_amdgcn_wmma_f32_16x16x32_bf16(
        false, a.v, false, b.v, (short)0, c, false, false);
}

__device__ __forceinline__ float redmax16(float v) {
    #pragma unroll
    for (int o = 1; o < 16; o <<= 1) v = fmaxf(v, __shfl_xor(v, o, 32));
    return v;
}
__device__ __forceinline__ float redsum16(float v) {
    #pragma unroll
    for (int o = 1; o < 16; o <<= 1) v += __shfl_xor(v, o, 32);
    return v;
}

// Raw split workgroup barrier (no implicit vmem fence -> keeps the
// double-buffered A loads in flight across the barrier). Async-LDS writes are
// already ordered by each thread's s_wait_asynccnt before signaling, and DS
// reads retire before the WMMAs that consume them, so no extra fence needed.
__device__ __forceinline__ void wg_barrier() {
    asm volatile("s_barrier_signal -1\n\t"
                 "s_barrier_wait -1" ::: "memory");
}

// ---------------------------------------------------------------------------
// LDS staging of a 128(n) x 32(k) bf16 B-tile (8 KB) per block k-step.
// Each of 256 threads moves 32 bytes via two async-to-LDS b128 ops; the
// instruction offset applies to BOTH the LDS and global address (ISA 10.x).
// ---------------------------------------------------------------------------
__device__ __forceinline__ void stage_b(const u16_t* __restrict__ Bt,
                                        int nblk, int k0,
                                        u16_t (*bs)[32], int tid) {
    int row  = tid >> 1;            // 0..127
    int half = (tid & 1) * 16;      // element offset within the 32-elem row
    const u16_t* g = Bt + (size_t)(nblk + row) * KDIM + k0 + half;
#if USE_ASYNC_LDS
    unsigned           lds = (unsigned)(unsigned long long)(void*)&bs[row][half];
    unsigned long long ga  = (unsigned long long)g;
    asm volatile(
        "global_load_async_to_lds_b128 %0, %1, off\n\t"
        "global_load_async_to_lds_b128 %0, %1, off offset:16"
        :: "v"(lds), "v"(ga) : "memory");
#else
    uint4 v0 = *(const uint4*)(g);
    uint4 v1 = *(const uint4*)(g + 8);
    *(uint4*)(&bs[row][half])     = v0;
    *(uint4*)(&bs[row][half + 8]) = v1;
#endif
}

__device__ __forceinline__ void wait_async_le2() {
#if USE_ASYNC_LDS
    asm volatile("s_wait_asynccnt 0x2" ::: "memory");
#endif
}
__device__ __forceinline__ void wait_async_le0() {
#if USE_ASYNC_LDS
    asm volatile("s_wait_asynccnt 0x0" ::: "memory");
#endif
}

// ---------------------------------------------------------------------------
// fp32 -> bf16 converts
// ---------------------------------------------------------------------------
__global__ void k_cvt(const float* __restrict__ s, u16_t* __restrict__ d, size_t n) {
    size_t i = (size_t)blockIdx.x * blockDim.x + threadIdx.x;
    size_t stride = (size_t)gridDim.x * blockDim.x;
    for (; i < n; i += stride) d[i] = f2bf(s[i]);
}

// dst[n*K + k] = src[k*srcLd + colOff + n]   (weight transpose + slice + cvt)
__global__ void k_cvt_t(const float* __restrict__ src, u16_t* __restrict__ dst,
                        int K, int srcLd, int colOff, size_t total) {
    size_t i = (size_t)blockIdx.x * blockDim.x + threadIdx.x;
    size_t stride = (size_t)gridDim.x * blockDim.x;
    for (; i < total; i += stride) {
        int n = (int)(i / K);
        int k = (int)(i % K);
        dst[i] = f2bf(src[(size_t)k * srcLd + colOff + n]);
    }
}

// ---------------------------------------------------------------------------
// Shared GEMM core: block tile 128x128 (8 waves as 4x2), wave tile 32x64.
// B staged in double-buffered LDS via async copies; A double-buffered in regs.
// All 4 B-frags loaded into independent registers before the 8-WMMA chain.
// ---------------------------------------------------------------------------
#define GEMM_CORE(A_, Bt_, ACC_)                                               \
    __shared__ __align__(16) u16_t Bs[2][128][32];                             \
    const int tid  = threadIdx.x;                                              \
    const int lane = tid & 31;                                                 \
    const int wv   = tid >> 5;                                                 \
    const int m0 = blockIdx.y * 128 + (wv >> 1) * 32;                          \
    const int nblk = blockIdx.x * 128;                                         \
    const int n0w  = (wv & 1) * 64;                                            \
    const v8f vz = {0,0,0,0,0,0,0,0};                                          \
    v8f ACC_[2][4];                                                            \
    _Pragma("unroll") for (int i = 0; i < 2; ++i)                              \
        _Pragma("unroll") for (int j = 0; j < 4; ++j) ACC_[i][j] = vz;         \
    stage_b(Bt_, nblk, 0, Bs[0], tid);                                         \
    Frag a0 = load_a(A_ + (size_t)m0 * KDIM, KDIM, lane);                      \
    Frag a1 = load_a(A_ + (size_t)(m0 + 16) * KDIM, KDIM, lane);               \
    for (int k0 = 0; k0 < KDIM; k0 += 32) {                                    \
        const int cur = (k0 >> 5) & 1;                                         \
        Frag an0, an1;                                                         \
        if (k0 + 32 < KDIM) {                                                  \
            stage_b(Bt_, nblk, k0 + 32, Bs[cur ^ 1], tid);                     \
            an0 = load_a(A_ + (size_t)m0 * KDIM + k0 + 32, KDIM, lane);        \
            an1 = load_a(A_ + (size_t)(m0 + 16) * KDIM + k0 + 32, KDIM, lane); \
            __builtin_prefetch(A_ + (size_t)m0 * KDIM + k0 + 64, 0, 3);        \
            wait_async_le2();                                                  \
        } else {                                                               \
            wait_async_le0();                                                  \
        }                                                                      \
        wg_barrier();                                                          \
        Frag b0 = load_bt(&Bs[cur][n0w +  0][0], 32, lane);                    \
        Frag b1 = load_bt(&Bs[cur][n0w + 16][0], 32, lane);                    \
        Frag b2 = load_bt(&Bs[cur][n0w + 32][0], 32, lane);                    \
        Frag b3 = load_bt(&Bs[cur][n0w + 48][0], 32, lane);                    \
        ACC_[0][0] = wmma_bf16(a0, b0, ACC_[0][0]);                            \
        ACC_[1][0] = wmma_bf16(a1, b0, ACC_[1][0]);                            \
        ACC_[0][1] = wmma_bf16(a0, b1, ACC_[0][1]);                            \
        ACC_[1][1] = wmma_bf16(a1, b1, ACC_[1][1]);                            \
        ACC_[0][2] = wmma_bf16(a0, b2, ACC_[0][2]);                            \
        ACC_[1][2] = wmma_bf16(a1, b2, ACC_[1][2]);                            \
        ACC_[0][3] = wmma_bf16(a0, b3, ACC_[0][3]);                            \
        ACC_[1][3] = wmma_bf16(a1, b3, ACC_[1][3]);                            \
        a0 = an0; a1 = an1;                                                    \
        wg_barrier();                                                          \
    }                                                                          \
    const int ml = lane & 15, hi = lane >> 4;

// ---------------------------------------------------------------------------
// GEMM 1: kv = x @ Wqkv[:, C:3C] + b  ->  K into [B*H][S][D], V into [B*H][D][S]
// ---------------------------------------------------------------------------
__global__ __launch_bounds__(256) void k_gemm_qkv(const u16_t* __restrict__ A,
                                                  const u16_t* __restrict__ Bt,
                                                  const float* __restrict__ bias,
                                                  u16_t* __restrict__ kbuf,
                                                  u16_t* __restrict__ vbufT) {
    GEMM_CORE(A, Bt, acc)
    #pragma unroll
    for (int i = 0; i < 2; ++i)
        #pragma unroll
        for (int j = 0; j < 4; ++j)
            #pragma unroll
            for (int r = 0; r < 8; ++r) {
                int row = m0 + 16 * i + r + hi * 8;
                int col = nblk + n0w + 16 * j + ml;
                float v = acc[i][j][r] + bias[col];
                int b = row >> 10, s = row & 1023;
                u16_t bv = f2bf(v);
                if (col < CN) {                       // K slice -> [B*H][S][D]
                    int h = col >> 6, d = col & 63;
                    kbuf[((size_t)(b * HN + h) * SN + s) * DN + d] = bv;
                } else {                              // V slice -> [B*H][D][S]
                    int c2 = col - CN;
                    int h = c2 >> 6, d = c2 & 63;
                    vbufT[((size_t)(b * HN + h) * DN + d) * SN + s] = bv;
                }
            }
}

// ---------------------------------------------------------------------------
// GEMM 2: out = attn_out @ Wproj + bproj   (fp32 output)
// ---------------------------------------------------------------------------
__global__ __launch_bounds__(256) void k_gemm_proj(const u16_t* __restrict__ A,
                                                   const u16_t* __restrict__ Bt,
                                                   const float* __restrict__ bias,
                                                   float* __restrict__ out) {
    GEMM_CORE(A, Bt, acc)
    #pragma unroll
    for (int i = 0; i < 2; ++i)
        #pragma unroll
        for (int j = 0; j < 4; ++j)
            #pragma unroll
            for (int r = 0; r < 8; ++r) {
                int row = m0 + 16 * i + r + hi * 8;
                int col = nblk + n0w + 16 * j + ml;
                out[(size_t)row * CN + col] = acc[i][j][r] + bias[col];
            }
}

// ---------------------------------------------------------------------------
// Flash attention: one wave per 16-row Q block (q = k = qkv[1]).
// Per 32-key block: 4 score WMMAs + online softmax + 4 P@V WMMAs.
// ---------------------------------------------------------------------------
__global__ __launch_bounds__(256) void k_attn(const u16_t* __restrict__ kb,
                                              const u16_t* __restrict__ vbT,
                                              u16_t* __restrict__ ab) {
    __shared__ __align__(16) u16_t Plds[8][16][32];

    const int lane = threadIdx.x & 31;
    const int wv   = threadIdx.x >> 5;
    const int gw   = blockIdx.x * 8 + wv;
    const int bh   = gw >> 6;          // b*H + h
    const int qb   = gw & 63;
    const int b    = bh >> 4;
    const int h    = bh & 15;
    const int q0   = qb * 16;
    const int ml   = lane & 15, hi = lane >> 4;

    const u16_t* Kp = kb  + (size_t)bh * SN * DN;   // [S][D]
    const u16_t* Vt = vbT + (size_t)bh * DN * SN;   // [D][S]

    Frag aq0 = load_a(Kp + (size_t)q0 * DN,      DN, lane);  // d 0..31
    Frag aq1 = load_a(Kp + (size_t)q0 * DN + 32, DN, lane);  // d 32..63

    const v8f vz = {0,0,0,0,0,0,0,0};
    v8f o[4]; o[0] = vz; o[1] = vz; o[2] = vz; o[3] = vz;
    float mrow[8], lrow[8];
    #pragma unroll
    for (int r = 0; r < 8; ++r) { mrow[r] = -1e30f; lrow[r] = 0.0f; }

    const float scale = 0.125f;        // 1/sqrt(64)
    const int nkb = (q0 + 47) >> 5;    // 32-key blocks covering keys <= q0+15

    for (int kb32 = 0; kb32 < nkb; ++kb32) {
        const int key0 = kb32 * 32;

        // scores for both 16-key sub-tiles; all 4 K-frags loaded up front
        Frag bk00 = load_bt(Kp + (size_t)(key0     ) * DN,      DN, lane);
        Frag bk01 = load_bt(Kp + (size_t)(key0     ) * DN + 32, DN, lane);
        Frag bk10 = load_bt(Kp + (size_t)(key0 + 16) * DN,      DN, lane);
        Frag bk11 = load_bt(Kp + (size_t)(key0 + 16) * DN + 32, DN, lane);

        v8f s[2];
        s[0] = wmma_bf16(aq0, bk00, vz);
        s[1] = wmma_bf16(aq0, bk10, vz);
        s[0] = wmma_bf16(aq1, bk01, s[0]);
        s[1] = wmma_bf16(aq1, bk11, s[1]);

        #pragma unroll
        for (int t = 0; t < 2; ++t)
            #pragma unroll
            for (int r = 0; r < 8; ++r) {
                int qr = q0 + r + hi * 8;
                int kc = key0 + 16 * t + ml;
                float x = s[t][r] * scale;
                s[t][r] = (kc <= qr) ? x : -1e30f;    // causal mask
            }

        float corr[8];
        #pragma unroll
        for (int r = 0; r < 8; ++r) {
            float v = redmax16(fmaxf(s[0][r], s[1][r]));
            float mn = fmaxf(mrow[r], v);
            corr[r] = __expf(mrow[r] - mn);
            mrow[r] = mn;
        }
        #pragma unroll
        for (int t = 0; t < 2; ++t)
            #pragma unroll
            for (int r = 0; r < 8; ++r)
                s[t][r] = __expf(s[t][r] - mrow[r]);
        #pragma unroll
        for (int r = 0; r < 8; ++r) {
            float sum = redsum16(s[0][r] + s[1][r]);
            lrow[r] = lrow[r] * corr[r] + sum;
        }
        #pragma unroll
        for (int j = 0; j < 4; ++j)
            #pragma unroll
            for (int r = 0; r < 8; ++r)
                o[j][r] *= corr[r];

        // C-layout -> A-layout remap through per-wave LDS tile (in-order DS ops)
        #pragma unroll
        for (int t = 0; t < 2; ++t)
            #pragma unroll
            for (int r = 0; r < 8; ++r)
                Plds[wv][r + hi * 8][t * 16 + ml] = f2bf(s[t][r]);

        Frag ap = load_a(&Plds[wv][0][0], 32, lane);

        // all 4 V-frags up front, then 4 WMMAs back-to-back
        Frag bv0 = load_bt(Vt + (size_t) 0 * SN + key0, SN, lane);
        Frag bv1 = load_bt(Vt + (size_t)16 * SN + key0, SN, lane);
        Frag bv2 = load_bt(Vt + (size_t)32 * SN + key0, SN, lane);
        Frag bv3 = load_bt(Vt + (size_t)48 * SN + key0, SN, lane);
        o[0] = wmma_bf16(ap, bv0, o[0]);
        o[1] = wmma_bf16(ap, bv1, o[1]);
        o[2] = wmma_bf16(ap, bv2, o[2]);
        o[3] = wmma_bf16(ap, bv3, o[3]);
    }

    // normalize + store to [B*S][C] (bf16) for the projection GEMM
    #pragma unroll
    for (int r = 0; r < 8; ++r) {
        int qr = q0 + r + hi * 8;
        float inv = 1.0f / lrow[r];
        size_t base = ((size_t)(b * SN + qr)) * CN + h * DN;
        #pragma unroll
        for (int j = 0; j < 4; ++j)
            ab[base + 16 * j + ml] = f2bf(o[j][r] * inv);
    }
}

// ---------------------------------------------------------------------------
// Launcher
// ---------------------------------------------------------------------------
extern "C" void kernel_launch(void* const* d_in, const int* in_sizes, int n_in,
                              void* d_out, int out_size, void* d_ws, size_t ws_size,
                              hipStream_t stream) {
    (void)in_sizes; (void)n_in; (void)out_size; (void)ws_size;

    const float* x     = (const float*)d_in[0];   // [B,S,C]
    const float* Wqkv  = (const float*)d_in[1];   // [C, 3C]
    const float* bqkv  = (const float*)d_in[2];   // [3C]
    const float* Wproj = (const float*)d_in[3];   // [C, C]
    const float* bproj = (const float*)d_in[4];   // [C]
    float* out = (float*)d_out;                   // [B,S,C] fp32

    const size_t XB_E  = (size_t)MROWS * KDIM;    // x in bf16
    const size_t WKV_E = (size_t)NKV   * KDIM;    // Wqkv[:,C:3C]^T bf16
    const size_t KB_E  = (size_t)BN * HN * SN * DN;
    const size_t VB_E  = KB_E;
    const size_t AB_E  = (size_t)MROWS * CN;
    const size_t WP_E  = (size_t)CN * CN;

    u16_t* p     = (u16_t*)d_ws;
    u16_t* xb    = p;  p += XB_E;
    u16_t* wkvb  = p;  p += WKV_E;
    u16_t* kb    = p;  p += KB_E;
    u16_t* vbT   = p;  p += VB_E;
    u16_t* ab    = p;  p += AB_E;
    u16_t* wpb   = p;  p += WP_E;

    k_cvt  <<<4096, 256, 0, stream>>>(x, xb, XB_E);
    k_cvt_t<<<4096, 256, 0, stream>>>(Wqkv,  wkvb, KDIM, 3 * CN, CN, WKV_E);
    k_cvt_t<<<2048, 256, 0, stream>>>(Wproj, wpb,  KDIM, CN,     0,  WP_E);

    k_gemm_qkv <<<dim3(NKV / 128, MROWS / 128), 256, 0, stream>>>(
        xb, wkvb, bqkv + CN, kb, vbT);

    k_attn <<<(BN * HN * (SN / 16)) / 8, 256, 0, stream>>>(kb, vbT, ab);

    k_gemm_proj<<<dim3(CN / 128, MROWS / 128), 256, 0, stream>>>(
        ab, wpb, bproj, out);
}
